// CapsuleNetwork_50654844289202
// MI455X (gfx1250) — matrix operand order repaired and checked by reading, not compiled, not run on hardware
//
#include <hip/hip_runtime.h>
#include <hip/hip_bf16.h>
#include <math.h>

typedef __attribute__((ext_vector_type(16))) _Float16 v16h;
typedef __attribute__((ext_vector_type(8)))  _Float16 v8h;
typedef __attribute__((ext_vector_type(8)))  unsigned short v8u;
typedef __attribute__((ext_vector_type(8)))  float    v8f;

// Problem constants
#define BATCH 512
#define N_IN  1152
#define D_IN  8
#define N_OUT 10
#define D_OUT 16
#define ND    160     // N_OUT * D_OUT

// ---------------------------------------------------------------------------
// Tensor Data Mover support (probe-confirmed builtin; arity differs by clang)
// ---------------------------------------------------------------------------
#if defined(__has_builtin)
#if __has_builtin(__builtin_amdgcn_tensor_load_to_lds) && \
    __has_builtin(__builtin_amdgcn_s_wait_tensorcnt)
#define HAS_TDM 1
#endif
#endif
#ifndef HAS_TDM
#define HAS_TDM 0
#endif

#if HAS_TDM
typedef __attribute__((ext_vector_type(4))) unsigned int u32x4;
typedef __attribute__((ext_vector_type(8))) int i32x8;
typedef __attribute__((ext_vector_type(4))) int i32x4;

// DMA a [rows x 32] f16 tile (row stride lda elements) from global into LDS,
// with TDM padding 16 DWORDs data + 8 DWORDs pad == LDS row stride 48 f16.
__device__ inline void tdm_load_tile_a(const _Float16* g, void* ldsPtr,
                                       int rows, int lda, int M) {
  unsigned long long ga = (unsigned long long)(const void*)g;
  unsigned lo = (unsigned)(size_t)ldsPtr;
  u32x4 g0;
  g0[0] = 1u;                                                 // count=1, user D#
  g0[1] = lo;                                                 // lds_addr (bytes)
  g0[2] = (unsigned)ga;                                       // global_addr[31:0]
  g0[3] = ((unsigned)(ga >> 32) & 0x01FFFFFFu) | (2u << 30);  // addr[56:32],type=2
  i32x8 g1;
  g1[0] = (1 << 16) | (1 << 20) | (3 << 22) | (7 << 25);  // 2B elems, pad 16dw->+8dw
  g1[1] = (lda & 0xFFFF) << 16;                           // tensor_dim0[15:0]
  g1[2] = ((lda >> 16) & 0xFFFF) | ((M & 0xFFFF) << 16);  // dim0 hi / dim1 lo
  g1[3] = ((M >> 16) & 0xFFFF) | (32 << 16);              // dim1 hi / tile_dim0=32
  g1[4] = rows & 0xFFFF;                                  // tile_dim1 (tile_dim2=0)
  g1[5] = lda;                                            // tensor_dim0_stride lo32
  g1[6] = 0;
  g1[7] = 0;
  i32x4 z4 = {0, 0, 0, 0};
#if __clang_major__ >= 23
  i32x8 z8 = {0, 0, 0, 0, 0, 0, 0, 0};
  __builtin_amdgcn_tensor_load_to_lds(g0, g1, z4, z4, z8, 0);
#else
  __builtin_amdgcn_tensor_load_to_lds(g0, g1, z4, z4, 0);
#endif
}
#endif

__device__ inline v8h v8h_zero() {
  v8h z;
#pragma unroll
  for (int j = 0; j < 8; ++j) z[j] = (_Float16)0.f;
  return z;
}

// ---------------------------------------------------------------------------
// Weight packing kernels (fp32 -> f16, K-major B matrices for WMMA)
// ---------------------------------------------------------------------------

// conv1 weights (256,1,9,9) -> B1[k=ky*9+kx (pad 96)][co] f16
__global__ void k_pack_b1(const float* __restrict__ w, _Float16* __restrict__ o) {
  int t = blockIdx.x * 256 + threadIdx.x;
  if (t >= 96 * 256) return;
  int k = t / 256, c = t % 256;
  o[t] = (k < 81) ? (_Float16)w[c * 81 + k] : (_Float16)0.f;
}

// conv2 weights (256,256,9,9) -> B2[k=(ky*9+kx)*256+ci][co] f16
__global__ void k_pack_b2(const float* __restrict__ w, _Float16* __restrict__ o) {
  int t = blockIdx.x * 256 + threadIdx.x;
  if (t >= 20736 * 256) return;
  int k = t / 256, co = t % 256;
  int p = k >> 8, ci = k & 255;
  o[t] = (_Float16)w[(size_t)(co * 256 + ci) * 81 + p];
}

// plain fp32 -> f16 convert (capsW)
__global__ void k_cvt_f16(const float* __restrict__ a, _Float16* __restrict__ o, int n) {
  int t = blockIdx.x * 256 + threadIdx.x;
  if (t < n) o[t] = (_Float16)a[t];
}

// transpose-convert: W[N,K] fp32 -> WT[K,N] f16
__global__ void k_packT(const float* __restrict__ w, _Float16* __restrict__ o, int K, int N) {
  int t = blockIdx.x * 256 + threadIdx.x;
  if (t >= K * N) return;
  int k = t / N, n = t % N;
  o[t] = (_Float16)w[(size_t)n * K + k];
}

// ---------------------------------------------------------------------------
// Generic WMMA f16 GEMM: C[M,N] = A[M,K] * B[K,N] (+bias, epilogue)
//   Block 256 threads = 8 waves, tile BM=64 x BN=128, BK=32, double-buffered.
//   AMODE: 0 = row-major f16 A (lda)  [A tile staged via TDM when available],
//          1 = conv1 im2col from f32 x, 2 = conv2 im2col from NHWC f16 h1
//   EPI:   0 = none, 1 = relu, 2 = sigmoid     (bias added first if non-null)
//   OMODE: 0 = f32 row-major, 1 = f16 row-major, 2 = capsule scatter f16
// ---------------------------------------------------------------------------
template <int AMODE, int EPI, int OMODE>
__global__ __launch_bounds__(256) void k_gemm(
    const _Float16* __restrict__ A, const _Float16* __restrict__ Bm,
    const float* __restrict__ bias, void* __restrict__ Cout,
    const float* __restrict__ Xf32, int M, int N, int K, int lda) {
  __shared__ _Float16 Al[2][64 * 48];    // A tiles 64x32, stride 48
  __shared__ _Float16 Bt[2][128 * 48];   // B tiles transposed 128x32, stride 48

  const int tid  = threadIdx.x;
  const int lane = tid & 31;
  const int wid  = tid >> 5;
  const int waveM = wid & 3;             // 4 row-groups of 16
  const int waveN = wid >> 2;            // 2 col-groups of 64
  const int half = lane >> 4;
  const int l16  = lane & 15;
  const int m0 = blockIdx.y * 64;
  const int n0 = blockIdx.x * 128;

  // A-stage mapping: 1 chunk of 8 f16 per thread (64 rows x 32 k)
  const int sRow = tid >> 2;
  const int sKK  = (tid & 3) * 8;
  // B-stage mapping: 2 adjacent K-rows x 8 cols per thread (32 k x 128 n)
  const int bKK = (tid >> 4) * 2;
  const int bNB = (tid & 15) * 8;

  const unsigned short* Bu = (const unsigned short*)Bm;

  auto loadAreg = [&](int k0) -> v8h {
    int grow = m0 + sRow;
    v8h av;
    if (AMODE == 0) {
      if (grow < M) av = *(const v8h*)(A + (size_t)grow * lda + k0 + sKK);
      else          av = v8h_zero();
    } else if (AMODE == 1) {
      // conv1: row -> (b,oy,ox) of 20x20; k -> (ky,kx) of 9x9, K padded to 96
      int b = grow / 400, rem = grow % 400;
      int oy = rem / 20, ox = rem % 20;
#pragma unroll
      for (int j = 0; j < 8; ++j) {
        int k = k0 + sKK + j;
        float val = 0.f;
        if (k < 81) {
          int ky = k / 9, kx = k % 9;
          val = Xf32[(size_t)b * 784 + (oy + ky) * 28 + (ox + kx)];
        }
        av[j] = (_Float16)val;
      }
    } else {
      // conv2: row -> (b,oy,ox) of 6x6; k -> (p=ky*9+kx, ci); h1 is NHWC f16
      int b = grow / 36, rem = grow % 36;
      int oy = rem / 6, ox = rem % 6;
      int k = k0 + sKK;
      int p = k >> 8, ci = k & 255;
      int iy = 2 * oy + p / 9, ix = 2 * ox + p % 9;
      av = *(const v8h*)(A + ((size_t)((b * 20 + iy) * 20 + ix)) * 256 + ci);
    }
    return av;
  };
  auto loadBreg = [&](int k0, v8u& r0, v8u& r1) {
    int gcol = n0 + bNB;
    if (gcol + 8 <= N) {
      r0 = *(const v8u*)(Bu + (size_t)(k0 + bKK) * N + gcol);
      r1 = *(const v8u*)(Bu + (size_t)(k0 + bKK + 1) * N + gcol);
    } else {
#pragma unroll
      for (int j = 0; j < 8; ++j) {
        r0[j] = (gcol + j < N) ? Bu[(size_t)(k0 + bKK) * N + gcol + j] : 0;
        r1[j] = (gcol + j < N) ? Bu[(size_t)(k0 + bKK + 1) * N + gcol + j] : 0;
      }
    }
  };
  auto storeB = [&](int buf, v8u r0, v8u r1) {
#pragma unroll
    for (int j = 0; j < 8; ++j)
      *(unsigned int*)(&Bt[buf][(bNB + j) * 48 + bKK]) =
          (unsigned int)r0[j] | ((unsigned int)r1[j] << 16);
  };

#if HAS_TDM
  constexpr bool useTdm = (AMODE == 0);
#else
  constexpr bool useTdm = false;
#endif

  v8f acc[4] = {};
  const int nk = K >> 5;   // all K are multiples of 32

  // ---- stage tile 0 ----
  {
#if HAS_TDM
    if (useTdm) {
      if (tid == 0)
        tdm_load_tile_a(A + (size_t)m0 * lda, &Al[0][0], 64, lda, M);
    }
#endif
    if (!useTdm) {
      v8h av = loadAreg(0);
      *(v8h*)(&Al[0][sRow * 48 + sKK]) = av;
    }
    v8u b0, b1;
    loadBreg(0, b0, b1);
    storeB(0, b0, b1);
#if HAS_TDM
    if (useTdm && tid < 32) __builtin_amdgcn_s_wait_tensorcnt(0);
#endif
  }
  __syncthreads();

  for (int kt = 0; kt < nk; ++kt) {
    const int cur = kt & 1, nxt = cur ^ 1;
    const bool more = (kt + 1) < nk;
    const int k0n = (kt + 1) << 5;

    // issue next-tile loads early (overlap with WMMA on current tile)
    v8h avN;
    v8u b0N, b1N;
#if HAS_TDM
    if (useTdm) {
      if (more && tid == 0)
        tdm_load_tile_a(A + (size_t)m0 * lda + k0n, &Al[nxt][0], 64, lda, M);
    }
#endif
    if (more) {
      if (!useTdm) avN = loadAreg(k0n);
      loadBreg(k0n, b0N, b1N);
    }

    // ---- compute current tile ----
    const _Float16* Ab = &Al[cur][0];
    const _Float16* Bb = &Bt[cur][0];
    v16h a;
    {
      int arow = waveM * 16 + l16;
      const v8h lo = *(const v8h*)(Ab + arow * 48 + 8 * half);       // K=8h..8h+7
      const v8h hi = *(const v8h*)(Ab + arow * 48 + 16 + 8 * half);  // K=16+8h..
#pragma unroll
      for (int j = 0; j < 8; ++j) { a[j] = lo[j]; a[8 + j] = hi[j]; }
    }
#pragma unroll
    for (int sub = 0; sub < 4; ++sub) {
      int ncol = waveN * 64 + sub * 16 + l16;
      v16h b;
      const v8h lo = *(const v8h*)(Bb + ncol * 48 + 8 * half);
      const v8h hi = *(const v8h*)(Bb + ncol * 48 + 16 + 8 * half);
#pragma unroll
      for (int j = 0; j < 8; ++j) { b[j] = lo[j]; b[8 + j] = hi[j]; }
      acc[sub] = __builtin_amdgcn_wmma_f32_16x16x32_f16(
          false, a, false, b, (short)0, acc[sub], false, false);
    }

    // ---- commit next tile to the other buffer ----
    if (more) {
      if (!useTdm) *(v8h*)(&Al[nxt][sRow * 48 + sKK]) = avN;
      storeB(nxt, b0N, b1N);
    }
#if HAS_TDM
    if (useTdm && more && tid < 32) __builtin_amdgcn_s_wait_tensorcnt(0);
#endif
    __syncthreads();
  }

  // ---- epilogue: C VGPR r -> row = r + 8*half, col = lane%16 ----
#pragma unroll
  for (int sub = 0; sub < 4; ++sub) {
    int col = n0 + waveN * 64 + sub * 16 + l16;
    if (col >= N) continue;
    float bb = bias ? bias[col] : 0.f;
#pragma unroll
    for (int r = 0; r < 8; ++r) {
      int row = m0 + waveM * 16 + r + 8 * half;
      if (row >= M) continue;
      float v = acc[sub][r] + bb;
      if (EPI == 1) v = v > 0.f ? v : 0.f;
      else if (EPI == 2) v = 1.f / (1.f + __expf(-v));
      if (OMODE == 0) {
        ((float*)Cout)[(size_t)row * N + col] = v;
      } else if (OMODE == 1) {
        ((_Float16*)Cout)[(size_t)row * N + col] = (_Float16)v;
      } else {
        // conv2 scatter: (b,oy,ox) x channel c=(g,i) -> u_pre[b][g*36+oy*6+ox][i]
        int b = row / 36, rem = row % 36;
        int g = col >> 3, i2 = col & 7;
        ((_Float16*)Cout)[((size_t)b * 1152 + g * 36 + rem) * 8 + i2] = (_Float16)v;
      }
    }
  }
}

// ---------------------------------------------------------------------------
// squash primary capsules: u_pre f16 [B,1152,8] -> u f16 [B,1152,8]
// ---------------------------------------------------------------------------
__global__ void k_squash(const _Float16* __restrict__ up, _Float16* __restrict__ u) {
  int t = blockIdx.x * 256 + threadIdx.x;
  if (t >= BATCH * N_IN) return;
  const v8h v = *(const v8h*)(up + (size_t)t * 8);
  float x[8], n2 = 0.f;
#pragma unroll
  for (int j = 0; j < 8; ++j) { x[j] = (float)v[j]; n2 += x[j] * x[j]; }
  float sc = n2 / (1.f + n2) * rsqrtf(fmaxf(n2, 1e-30f));
  v8h o;
#pragma unroll
  for (int j = 0; j < 8; ++j) o[j] = (_Float16)(x[j] * sc);
  *(v8h*)(u + (size_t)t * 8) = o;
}

// ---------------------------------------------------------------------------
// capsule transform: per n, GEMM [B=512 x 8] x [8 x 160] (K padded 8->32)
// grid = (1152 capsules, 32 M-tiles, 10 N-tiles), one wave32 per 16x16 tile
// ---------------------------------------------------------------------------
__global__ __launch_bounds__(32) void k_caps(
    const _Float16* __restrict__ u, const _Float16* __restrict__ cw,
    _Float16* __restrict__ uhat) {
  const int n  = blockIdx.x;
  const int mt = blockIdx.y;
  const int nt = blockIdx.z;
  const int lane = threadIdx.x;
  const int half = lane >> 4, l16 = lane & 15;

  v16h a, b;
#pragma unroll
  for (int j = 0; j < 16; ++j) { a[j] = (_Float16)0.f; b[j] = (_Float16)0.f; }

  if (half == 0) {
    int mrow = mt * 16 + l16;                                  // batch row
    const v8h av = *(const v8h*)(u + ((size_t)mrow * N_IN + n) * 8);
#pragma unroll
    for (int j = 0; j < 8; ++j) a[j] = av[j];                  // K = 0..7 real
    int col = nt * 16 + l16;                                   // output col
#pragma unroll
    for (int j = 0; j < 8; ++j) b[j] = cw[(size_t)n * 1280 + j * 160 + col];
  }
  v8f acc = {};
  acc = __builtin_amdgcn_wmma_f32_16x16x32_f16(false, a, false, b, (short)0,
                                               acc, false, false);
  int col = nt * 16 + l16;
#pragma unroll
  for (int r = 0; r < 8; ++r) {
    int row = mt * 16 + r + 8 * half;
    uhat[((size_t)row * N_IN + n) * ND + col] = (_Float16)acc[r];
  }
}

// ---------------------------------------------------------------------------
// dynamic routing (3 iters), one block per batch element, state in LDS
// writes pred -> dout_pred[b*10+o], v -> dout_v[b*160+od]
// ---------------------------------------------------------------------------
__global__ __launch_bounds__(256) void k_rout(const _Float16* __restrict__ uhat,
                                              float* __restrict__ dout_pred,
                                              float* __restrict__ dout_v) {
  extern __shared__ float sm[];
  float* blog = sm;               // [1152*10]
  float* cbuf = sm + 11520;       // [1152*10]
  float* sv   = sm + 23040;       // [160]
  float* vv   = sm + 23200;       // [160]
  float* n2s  = sm + 23360;       // [16]

  const int b = blockIdx.x;
  const int t = threadIdx.x;
  const _Float16* U = uhat + (size_t)b * N_IN * ND;

  for (int i = t; i < N_IN * N_OUT; i += 256) blog[i] = 0.f;
  __syncthreads();

  for (int it = 0; it < 3; ++it) {
    // c = softmax(b) over nOut
    for (int n = t; n < N_IN; n += 256) {
      float mx = -1e30f;
#pragma unroll
      for (int o = 0; o < N_OUT; ++o) mx = fmaxf(mx, blog[n * N_OUT + o]);
      float e[N_OUT], ssum = 0.f;
#pragma unroll
      for (int o = 0; o < N_OUT; ++o) {
        e[o] = __expf(blog[n * N_OUT + o] - mx);
        ssum += e[o];
      }
      float inv = 1.f / ssum;
#pragma unroll
      for (int o = 0; o < N_OUT; ++o) cbuf[n * N_OUT + o] = e[o] * inv;
    }
    __syncthreads();
    // s[o,d] = sum_n c[n,o] * uhat[n,o,d]
    if (t < ND) {
      int o = t >> 4;
      float acc = 0.f;
      for (int n = 0; n < N_IN; ++n)
        acc += cbuf[n * N_OUT + o] * (float)U[(size_t)n * ND + t];
      sv[t] = acc;
    }
    __syncthreads();
    // v = squash(s)
    if (t < N_OUT) {
      float n2 = 0.f;
#pragma unroll
      for (int d = 0; d < D_OUT; ++d) { float x = sv[t * 16 + d]; n2 += x * x; }
      float sc = n2 / (1.f + n2) * rsqrtf(fmaxf(n2, 1e-30f));
#pragma unroll
      for (int d = 0; d < D_OUT; ++d) vv[t * 16 + d] = sv[t * 16 + d] * sc;
      n2s[t] = n2;
    }
    __syncthreads();
    // agreement: b += uhat . v
    if (it < 2) {
      for (int n = t; n < N_IN; n += 256) {
#pragma unroll
        for (int o = 0; o < N_OUT; ++o) {
          float acc = 0.f;
#pragma unroll
          for (int d = 0; d < D_OUT; ++d)
            acc += (float)U[(size_t)n * ND + o * 16 + d] * vv[o * 16 + d];
          blog[n * N_OUT + o] += acc;
        }
      }
      __syncthreads();
    }
  }
  if (t < N_OUT) dout_pred[b * N_OUT + t] = n2s[t] / (1.f + n2s[t]);  // |v|
  if (t < ND)    dout_v[b * ND + t] = vv[t];
}

// ---------------------------------------------------------------------------
// decoder input: act = v masked by true label, f16
// ---------------------------------------------------------------------------
__global__ void k_act(const float* __restrict__ vout, const int* __restrict__ labels,
                      _Float16* __restrict__ act) {
  int t = blockIdx.x * 256 + threadIdx.x;
  if (t >= BATCH * ND) return;
  int b = t / ND, od = t % ND, o = od >> 4;
  act[t] = (labels[b] == o) ? (_Float16)vout[t] : (_Float16)0.f;
}

// ---------------------------------------------------------------------------
extern "C" void kernel_launch(void* const* d_in, const int* in_sizes, int n_in,
                              void* d_out, int out_size, void* d_ws, size_t ws_size,
                              hipStream_t stream) {
  (void)in_sizes; (void)n_in; (void)out_size; (void)ws_size;
  const float* x       = (const float*)d_in[0];
  const int*   labels  = (const int*)d_in[1];
  const float* conv1_w = (const float*)d_in[2];
  const float* conv1_b = (const float*)d_in[3];
  const float* conv2_w = (const float*)d_in[4];
  const float* conv2_b = (const float*)d_in[5];
  const float* capsW   = (const float*)d_in[6];
  const float* fc1_w   = (const float*)d_in[7];
  const float* fc1_b   = (const float*)d_in[8];
  const float* fc2_w   = (const float*)d_in[9];
  const float* fc2_b   = (const float*)d_in[10];
  const float* fc3_w   = (const float*)d_in[11];
  const float* fc3_b   = (const float*)d_in[12];

  char* ws = (char*)d_ws;
  size_t off = 0;
  auto alloc = [&](size_t bytes) -> char* {
    char* p = ws + off;
    off = (off + bytes + 255) & ~(size_t)255;
    return p;
  };

  // h1 (NHWC f16, 104.9 MB) and uhat (f16, 188.7 MB) alias: disjoint lifetimes
  _Float16* h1   = (_Float16*)ws;
  _Float16* uhat = (_Float16*)ws;
  off = ((size_t)BATCH * N_IN * ND * 2 + 255) & ~(size_t)255;

  _Float16* up   = (_Float16*)alloc((size_t)BATCH * N_IN * D_IN * 2);
  _Float16* u16  = (_Float16*)alloc((size_t)BATCH * N_IN * D_IN * 2);
  _Float16* B1   = (_Float16*)alloc(96 * 256 * 2);
  _Float16* B2   = (_Float16*)alloc((size_t)20736 * 256 * 2);
  _Float16* CW   = (_Float16*)alloc((size_t)N_IN * D_IN * ND * 2);
  _Float16* F1   = (_Float16*)alloc(160 * 512 * 2);
  _Float16* F2   = (_Float16*)alloc(512 * 1024 * 2);
  _Float16* F3   = (_Float16*)alloc(1024 * 784 * 2);
  _Float16* ACT  = (_Float16*)alloc((size_t)BATCH * ND * 2);
  _Float16* R1   = (_Float16*)alloc((size_t)BATCH * 512 * 2);
  _Float16* R2   = (_Float16*)alloc((size_t)BATCH * 1024 * 2);

  float* out    = (float*)d_out;
  float* o_pred = out;                  // [512,10]
  float* o_rec  = out + 5120;           // [512,784]
  float* o_v    = out + 5120 + 401408;  // [512,160]

  // ---- pack weights ----
  k_pack_b1<<<(96 * 256 + 255) / 256, 256, 0, stream>>>(conv1_w, B1);
  k_pack_b2<<<(20736 * 256 + 255) / 256, 256, 0, stream>>>(conv2_w, B2);
  k_cvt_f16<<<(N_IN * D_IN * ND + 255) / 256, 256, 0, stream>>>(capsW, CW, N_IN * D_IN * ND);
  k_packT<<<(160 * 512 + 255) / 256, 256, 0, stream>>>(fc1_w, F1, 160, 512);
  k_packT<<<(512 * 1024 + 255) / 256, 256, 0, stream>>>(fc2_w, F2, 512, 1024);
  k_packT<<<(1024 * 784 + 255) / 256, 256, 0, stream>>>(fc3_w, F3, 1024, 784);

  // ---- conv1 + relu -> h1 NHWC f16 : M=512*400, N=256, K=96 (81 padded) ----
  k_gemm<1, 1, 1><<<dim3(2, 3200), 256, 0, stream>>>(
      nullptr, B1, conv1_b, h1, x, BATCH * 400, 256, 96, 0);

  // ---- conv2 (implicit GEMM) -> u_pre scatter : M=512*36, N=256, K=20736 ----
  k_gemm<2, 0, 2><<<dim3(2, 288), 256, 0, stream>>>(
      h1, B2, conv2_b, up, nullptr, BATCH * 36, 256, 20736, 0);

  // ---- squash primary capsules ----
  k_squash<<<(BATCH * N_IN + 255) / 256, 256, 0, stream>>>(up, u16);

  // ---- uhat = u x capsW (batched over 1152 capsules) ----
  k_caps<<<dim3(N_IN, BATCH / 16, N_OUT), 32, 0, stream>>>(u16, CW, uhat);

  // ---- dynamic routing: pred + v straight into d_out ----
  size_t routLds = (size_t)(23360 + 16) * sizeof(float);
  k_rout<<<BATCH, 256, routLds, stream>>>(uhat, o_pred, o_v);

  // ---- decoder (A tiles staged by the Tensor Data Mover when available) ----
  k_act<<<(BATCH * ND + 255) / 256, 256, 0, stream>>>(o_v, labels, ACT);
  k_gemm<0, 1, 1><<<dim3(4, BATCH / 64), 256, 0, stream>>>(
      ACT, F1, fc1_b, R1, nullptr, BATCH, 512, 160, 160);
  k_gemm<0, 1, 1><<<dim3(8, BATCH / 64), 256, 0, stream>>>(
      R1, F2, fc2_b, R2, nullptr, BATCH, 1024, 512, 512);
  k_gemm<0, 2, 0><<<dim3(7, BATCH / 64), 256, 0, stream>>>(
      R2, F3, fc3_b, o_rec, nullptr, BATCH, 784, 1024, 1024);
}
